// TypeNet_30124900614422
// MI455X (gfx1250) — matrix operand order repaired and checked by reading, not compiled, not run on hardware
//
#include <hip/hip_runtime.h>
#include <hip/hip_bf16.h>

// CDNA5 / gfx1250 implementation. wave32. WMMA f32_16x16x32_f16 used for:
//   (a) the per-step recurrent matvec h @ Whh^T inside the sequential LSTM scans
//   (b) the parallel input-projection GEMM (xbn2 @ Wih2^T) with BN fused into A
// The workload is latency-bound on 2x2048 sequential LSTM steps per branch, so
// the scan kernel keeps Whh (f16) register-resident, broadcasts h via LDS, and
// prefetches the next step's x-projection into registers.

typedef __attribute__((ext_vector_type(16))) _Float16 v16h;  // 32B: WMMA A/B operand
typedef __attribute__((ext_vector_type(8)))  _Float16 v8h;   // 16B half-chunk
typedef __attribute__((ext_vector_type(8)))  float    v8f;   // WMMA C/D accumulator

#define EPS 1e-5f

// ---------------------------------------------------------------- utilities

__global__ __launch_bounds__(256) void f32_to_f16_kernel(
    const float* __restrict__ src, _Float16* __restrict__ dst, int n) {
  int i = blockIdx.x * 256 + threadIdx.x;
  if (i < n) dst[i] = (_Float16)src[i];
}

// BN1 stats: x is [128,48,3]; branch br uses channels 16*br..16*br+15 of dim1.
// stats over N=128, L=3 (384 values / channel). grid=48 (br*16+c), block=128.
__global__ __launch_bounds__(128) void bn1_stats_kernel(
    const float* __restrict__ x, float* __restrict__ stats1) {
  int br = blockIdx.x >> 4, c = blockIdx.x & 15, tid = threadIdx.x;
  const float* p = x + tid * 144 + (16 * br + c) * 3;
  float s = p[0] + p[1] + p[2];
  float q = p[0] * p[0] + p[1] * p[1] + p[2] * p[2];
  __shared__ float ss[128], sq[128];
  ss[tid] = s; sq[tid] = q; __syncthreads();
  for (int o = 64; o > 0; o >>= 1) {
    if (tid < o) { ss[tid] += ss[tid + o]; sq[tid] += sq[tid + o]; }
    __syncthreads();
  }
  if (tid == 0) {
    float m = ss[0] / 384.f;
    float v = sq[0] / 384.f - m * m;
    stats1[blockIdx.x * 2]     = m;
    stats1[blockIdx.x * 2 + 1] = rsqrtf(v + EPS);
  }
}

// xproj1[br][s][g] = (bih1+bhh1)[g] + sum_d bn1(x)[s,d] * Wih1[g,d]
// s = t*128 + b (t = channel of BN1). K=3 -> plain VALU is optimal here.
__global__ __launch_bounds__(256) void xproj1_kernel(
    const float* __restrict__ x, const float* __restrict__ Wih1,
    const float* __restrict__ bih1, const float* __restrict__ bhh1,
    const float* __restrict__ stats1, const float* __restrict__ g1,
    const float* __restrict__ b1, float* __restrict__ xproj1) {
  int idx = blockIdx.x * 256 + threadIdx.x;  // < 3 * 2^20
  int g  = idx & 511;
  int s  = (idx >> 9) & 2047;
  int br = idx >> 20;
  int t = s >> 7, b = s & 127;
  float mean = stats1[(br * 16 + t) * 2];
  float rstd = stats1[(br * 16 + t) * 2 + 1];
  float scale = g1[t] * rstd;
  float shift = b1[t] - mean * scale;
  const float* xr = x + b * 144 + (16 * br + t) * 3;
  const float* wr = Wih1 + g * 3;
  float v = bih1[g] + bhh1[g]
          + (xr[0] * scale + shift) * wr[0]
          + (xr[1] * scale + shift) * wr[1]
          + (xr[2] * scale + shift) * wr[2];
  xproj1[((size_t)br << 20) + (size_t)s * 512 + g] = v;
}

// BN2 stats over ys1 viewed as [16,128,128] per branch: channel c = dim1,
// stats over n<16, l<128 (2048 values). grid = 384 (br*128+c), block = 256.
__global__ __launch_bounds__(256) void bn2_stats_kernel(
    const float* __restrict__ ys1, float* __restrict__ stats2) {
  int br = blockIdx.x >> 7, c = blockIdx.x & 127, tid = threadIdx.x;
  const float* base = ys1 + (size_t)br * 2048 * 128;
  float s = 0.f, q = 0.f;
#pragma unroll
  for (int i = 0; i < 8; ++i) {
    int idx = tid + 256 * i;           // 0..2047
    int n = idx >> 7, l = idx & 127;
    float v = base[(size_t)(n * 128 + c) * 128 + l];
    s += v; q += v * v;
  }
  __shared__ float ss[256], sq[256];
  ss[tid] = s; sq[tid] = q; __syncthreads();
  for (int o = 128; o > 0; o >>= 1) {
    if (tid < o) { ss[tid] += ss[tid + o]; sq[tid] += sq[tid + o]; }
    __syncthreads();
  }
  if (tid == 0) {
    float m = ss[0] / 2048.f;
    float v = sq[0] / 2048.f - m * m;
    stats2[blockIdx.x * 2]     = m;
    stats2[blockIdx.x * 2 + 1] = rsqrtf(v + EPS);
  }
}

// ---------------------------------------------------- xproj2 GEMM via WMMA
// xproj2[br][s2][g] = (bih2+bhh2)[g] + sum_d xbn2[s2][d] * Wih2[g][d]
// s2 = c*16 + n; xbn2[s2][d] = BN2(ys1[n*128+c][d]), channel c = s2>>4.
// M=2048 N=512 K=128 per branch; one 16x16 tile per wave, 4 k-step WMMAs.
__global__ __launch_bounds__(256) void xproj2_gemm_kernel(
    const float* __restrict__ ys1, const _Float16* __restrict__ Wih2h,
    const float* __restrict__ stats2, const float* __restrict__ g2,
    const float* __restrict__ b2, const float* __restrict__ bih2,
    const float* __restrict__ bhh2, float* __restrict__ xproj2) {
  int br   = blockIdx.y;
  int tid  = threadIdx.x;
  int lane = tid & 31;
  int ln   = lane & 15;     // N-lane / M-row within tile
  int hw   = lane >> 4;     // half-wave select (K split per ISA layout)
  int id   = blockIdx.x * 8 + (tid >> 5);   // 0..4095
  int tm = id >> 5, tn = id & 31;

  int s2 = 16 * tm + ln;                       // A row this lane feeds
  int c = s2 >> 4, n = s2 & 15;
  const float* ysrow = ys1 + (size_t)br * 2048 * 128 + (size_t)(n * 128 + c) * 128;
  float rstd  = stats2[(br * 128 + c) * 2 + 1];
  float scale = g2[c] * rstd;
  float shift = b2[c] - stats2[(br * 128 + c) * 2] * scale;

  v8f acc = {};
#pragma unroll
  for (int kc = 0; kc < 4; ++kc) {
    // A 16x32 f16 layout: lane l elem e -> K = 32kc + 16*(l>>4) + e (matches B)
    const float* p0 = ysrow + 32 * kc + 16 * hw;
    const float* p1 = p0 + 8;
    union { v16h v; _Float16 e[16]; } ua;
#pragma unroll
    for (int j = 0; j < 8; ++j) {
      ua.e[j]     = (_Float16)(p0[j] * scale + shift);
      ua.e[8 + j] = (_Float16)(p1[j] * scale + shift);
    }
    // B 32x16 f16 layout: lane l elem e -> K = 32kc + 16*(l>>4) + e, N = l&15
    v16h bv = *(const v16h*)(Wih2h + (size_t)(16 * tn + ln) * 128 + 32 * kc + 16 * hw);
    acc = __builtin_amdgcn_wmma_f32_16x16x32_f16(
        false, ua.v, false, bv, (short)0, acc, false, false);
  }
  // D layout: VGPR r = row r (lanes<16) / row r+8 (lanes>=16), N = l&15.
  float* outb = xproj2 + (size_t)br * 2048 * 512;
#pragma unroll
  for (int r = 0; r < 8; ++r) {
    int srow = 16 * tm + r + 8 * hw;
    int g = 16 * tn + ln;
    outb[(size_t)srow * 512 + g] = acc[r] + bih2[g] + bhh2[g];
  }
}

// ------------------------------------------------------- sequential LSTM scan
// grid=3 (branch), block=512 (16 waves). Wave w owns gate rows [32w,32w+32).
// Whh (f16) register-resident; h replicated into all 16 A rows so every lane's
// D element 0 directly yields gates[tile*16 + (lane&15)].
__global__ __launch_bounds__(512) void lstm_scan_kernel(
    const float* __restrict__ xproj,      // [branch][2048][512] (bias included)
    const _Float16* __restrict__ Whh16,   // [512][128]
    const float* __restrict__ h0s, const float* __restrict__ c0s,  // [6][128]
    int layer, float* __restrict__ ys) {  // [branch][2048][128]
  int br = blockIdx.x;
  int tid = threadIdx.x;
  int lane = tid & 31;
  int w = tid >> 5;           // 0..15
  int ln = lane & 15;
  int hw = lane >> 4;
  int row0 = 32 * w;

  __shared__ __attribute__((aligned(16))) _Float16 hbuf[128];
  __shared__ float gbuf[512];

  const float* xp = xproj + (size_t)br * 2048 * 512;
  float* y = ys + (size_t)br * 2048 * 128;
  const float* h0 = h0s + (2 * br + layer) * 128;
  const float* c0 = c0s + (2 * br + layer) * 128;

  // Register-resident B tiles (Whh^T). B elem e @ lane l: K=32kc+16*hw+e, N=ln.
  v16h B[2][4];
#pragma unroll
  for (int t = 0; t < 2; ++t)
#pragma unroll
    for (int kc = 0; kc < 4; ++kc)
      B[t][kc] = *(const v16h*)(Whh16 + (size_t)(row0 + 16 * t + ln) * 128 + 32 * kc + 16 * hw);

  float cj = 0.f;
  if (tid < 128) { cj = c0[tid]; hbuf[tid] = (_Float16)h0[tid]; }
  __syncthreads();

  // prefetch x-projection for step 0
  float xi = 0.f, xf = 0.f, xg = 0.f, xo = 0.f;
  if (tid < 128) { xi = xp[tid]; xf = xp[128 + tid]; xg = xp[256 + tid]; xo = xp[384 + tid]; }

  for (int s = 0; s < 2048; ++s) {
    v8f acc0 = {}, acc1 = {};
#pragma unroll
    for (int kc = 0; kc < 4; ++kc) {
      // A rows all = h: lane elem e -> K = 32kc + 16*hw + e (matches B).
      union { v16h v; v8h h2[2]; } ua;
      ua.h2[0] = *(const v8h*)&hbuf[32 * kc + 16 * hw];
      ua.h2[1] = *(const v8h*)&hbuf[32 * kc + 16 * hw + 8];
      acc0 = __builtin_amdgcn_wmma_f32_16x16x32_f16(
          false, ua.v, false, B[0][kc], (short)0, acc0, false, false);
      acc1 = __builtin_amdgcn_wmma_f32_16x16x32_f16(
          false, ua.v, false, B[1][kc], (short)0, acc1, false, false);
    }
    if (hw == 0) {             // rows replicated -> lanes 16..31 are duplicates
      gbuf[row0 + ln]      = acc0[0];
      gbuf[row0 + 16 + ln] = acc1[0];
    }
    __syncthreads();
    if (tid < 128) {
      float gi = gbuf[tid]       + xi;
      float gf = gbuf[128 + tid] + xf;
      float gg = gbuf[256 + tid] + xg;
      float go = gbuf[384 + tid] + xo;
      float i = 1.f / (1.f + __expf(-gi));
      float f = 1.f / (1.f + __expf(-gf));
      float g = tanhf(gg);
      float o = 1.f / (1.f + __expf(-go));
      cj = f * cj + i * g;
      float h = o * tanhf(cj);
      y[(size_t)s * 128 + tid] = h;
      hbuf[tid] = (_Float16)h;
      if (s + 1 < 2048) {      // prefetch next step's projection off the chain
        const float* nx = xp + (size_t)(s + 1) * 512;
        xi = nx[tid]; xf = nx[128 + tid]; xg = nx[256 + tid]; xo = nx[384 + tid];
      }
    }
    __syncthreads();
  }
}

// ------------------------------------------------------------------- launch

extern "C" void kernel_launch(void* const* d_in, const int* in_sizes, int n_in,
                              void* d_out, int out_size, void* d_ws, size_t ws_size,
                              hipStream_t stream) {
  (void)in_sizes; (void)n_in; (void)out_size; (void)ws_size;
  const float* x    = (const float*)d_in[0];
  const float* g1   = (const float*)d_in[1];
  const float* b1   = (const float*)d_in[2];
  const float* Wih1 = (const float*)d_in[3];
  const float* Whh1 = (const float*)d_in[4];
  const float* bih1 = (const float*)d_in[5];
  const float* bhh1 = (const float*)d_in[6];
  const float* g2   = (const float*)d_in[7];
  const float* b2   = (const float*)d_in[8];
  const float* Wih2 = (const float*)d_in[9];
  const float* Whh2 = (const float*)d_in[10];
  const float* bih2 = (const float*)d_in[11];
  const float* bhh2 = (const float*)d_in[12];
  const float* h0s  = (const float*)d_in[13];
  const float* c0s  = (const float*)d_in[14];

  char* ws = (char*)d_ws;
  float*    stats1 = (float*)(ws + 0);                    //   384 B
  float*    stats2 = (float*)(ws + 512);                  //  3072 B
  float*    xproj1 = (float*)(ws + 4096);                 // 12 MB
  float*    ys1    = (float*)(ws + 4096 + 12582912);      //  3 MB
  float*    xproj2 = (float*)(ws + 4096 + 12582912 + 3145728);          // 12 MB
  _Float16* Whh1h  = (_Float16*)(ws + 4096 + 12582912 + 3145728 + 12582912);
  _Float16* Whh2h  = Whh1h + 65536;
  _Float16* Wih2h  = Whh2h + 65536;
  float* out = (float*)d_out;

  // weight conversions (parallel, tiny)
  f32_to_f16_kernel<<<256, 256, 0, stream>>>(Whh1, Whh1h, 65536);
  f32_to_f16_kernel<<<256, 256, 0, stream>>>(Whh2, Whh2h, 65536);
  f32_to_f16_kernel<<<256, 256, 0, stream>>>(Wih2, Wih2h, 65536);

  // branch pipeline (branches computed together; stream order gives deps)
  bn1_stats_kernel<<<48, 128, 0, stream>>>(x, stats1);
  xproj1_kernel<<<12288, 256, 0, stream>>>(x, Wih1, bih1, bhh1, stats1, g1, b1, xproj1);
  lstm_scan_kernel<<<3, 512, 0, stream>>>(xproj1, Whh1h, h0s, c0s, 0, ys1);
  bn2_stats_kernel<<<384, 256, 0, stream>>>(ys1, stats2);
  xproj2_gemm_kernel<<<dim3(512, 3), 256, 0, stream>>>(ys1, Wih2h, stats2, g2, b2,
                                                       bih2, bhh2, xproj2);
  lstm_scan_kernel<<<3, 512, 0, stream>>>(xproj2, Whh2h, h0s, c0s, 1, out);
}
